// GPT_70342974374193
// MI455X (gfx1250) — compile-verified, matrix operand order checked by reading, not run on hardware
//
#include <hip/hip_runtime.h>

typedef unsigned short ushort_t;
typedef unsigned int uint_t;

typedef __attribute__((ext_vector_type(16))) __bf16 v16bf;
typedef __attribute__((ext_vector_type(2)))  __bf16 v2bf;
typedef __attribute__((ext_vector_type(8)))  float  v8f;

struct alignas(16) U128 { uint_t x, y, z, w; };
union FragU { U128 q[2]; v16bf v; };

__device__ __forceinline__ ushort_t f2b(float x) {
  __bf16 h = (__bf16)x;                       // native v_cvt, RNE
  return __builtin_bit_cast(ushort_t, h);
}
__device__ __forceinline__ uint_t pack2(float a, float b) {
  v2bf t; t[0] = (__bf16)a; t[1] = (__bf16)b; // fuses to v_cvt_pk_bf16_f32
  return __builtin_bit_cast(uint_t, t);
}
__device__ __forceinline__ float b2f(ushort_t h) {
  return __uint_as_float(((uint_t)h) << 16);
}

__device__ __forceinline__ float blockReduceSum256(float v) {
  __shared__ float sm[9];
#pragma unroll
  for (int d = 16; d; d >>= 1) v += __shfl_xor(v, d, 32);
  if ((threadIdx.x & 31) == 0) sm[threadIdx.x >> 5] = v;
  __syncthreads();
  if (threadIdx.x == 0) {
    float t = 0.f;
    int nw = (blockDim.x + 31) >> 5;
    for (int i = 0; i < nw; ++i) t += sm[i];
    sm[8] = t;
  }
  __syncthreads();
  return sm[8];
}

// ---------------- embedding gather ----------------
__global__ __launch_bounds__(256)
void embed_kernel(const int* __restrict__ idx, const float* __restrict__ wte,
                  float* __restrict__ x) {
  int n = blockIdx.x;
  const float* src = wte + (size_t)idx[n] * 1024;
  float* dst = x + (size_t)n * 1024;
  for (int c = threadIdx.x; c < 1024; c += 256) dst[c] = src[c];
}

// ---------------- rmsnorm (fp32 in, bf16 and/or fp32 out) ----------------
__global__ __launch_bounds__(256)
void rmsnorm_kernel(const float* __restrict__ x, const float* __restrict__ w,
                    ushort_t* __restrict__ ob, float* __restrict__ of,
                    int rowScale, int rowOff) {
  int row = blockIdx.x * rowScale + rowOff;
  const float* xr = x + (size_t)row * 1024;
  float ss = 0.f;
  for (int c = threadIdx.x; c < 1024; c += 256) { float v = xr[c]; ss += v * v; }
  ss = blockReduceSum256(ss);
  float inv = rsqrtf(ss * (1.0f / 1024.0f) + 1e-5f);
  for (int c = threadIdx.x; c < 1024; c += 256) {
    float o = xr[c] * inv * w[c];
    if (ob) ob[(size_t)blockIdx.x * 1024 + c] = f2b(o);
    if (of) of[(size_t)blockIdx.x * 1024 + c] = o;
  }
}

// ---------------- generic bf16 WMMA GEMM (optionally expert-routed) -------
// C[base+m, :] (+)= A[rowmap(base+m), :] * bf16(B_e[:, :])
// 128x128 block tile, 8 waves (each 32x64 = 2x4 wmma tiles), KT=32.
// Double-buffered LDS; A tile streamed with global_load_async_to_lds_b128
// (ASYNCcnt), B tile converted fp32->bf16 through registers.
__global__ __launch_bounds__(256)
void gemm_bf16_kernel(const ushort_t* __restrict__ A, int lda,
                      const int* __restrict__ rowMap,
                      const float* __restrict__ B, int ldb, long long bStride,
                      float* __restrict__ C, int ldc, int addToC,
                      int M, int N, int K,
                      const int* __restrict__ eOff, const int* __restrict__ eCnt) {
  const int e = blockIdx.z;
  const int base = eOff ? eOff[e] : 0;
  const int cnt  = eCnt ? eCnt[e] : M;
  const int m0 = blockIdx.x * 128;
  if (m0 >= cnt) return;
  const int n0 = blockIdx.y * 128;
  const float* Bp = B + (long long)e * bStride;

  __shared__ __align__(16) ushort_t Al[2][128 * 32];   // [buf][m][k] bf16
  __shared__ __align__(16) ushort_t Bl[2][128 * 32];   // [buf][n][k] bf16

  const int tid  = threadIdx.x;
  const int lane = tid & 31;
  const int wid  = tid >> 5;
  const int wm   = wid & 3;   // 4 waves along M (32 rows each)
  const int wn   = wid >> 2;  // 2 waves along N (64 cols each)
  const int ln   = lane & 15;
  const int half = lane >> 4;

  v8f zero = {0.f,0.f,0.f,0.f,0.f,0.f,0.f,0.f};
  v8f acc[2][4];
#pragma unroll
  for (int i = 0; i < 2; ++i)
#pragma unroll
    for (int j = 0; j < 4; ++j) acc[i][j] = zero;

  // A staging: thread -> (row, 16-col half-row)
  const int ldRow = tid >> 1;          // 0..127
  const int ldCol = (tid & 1) * 16;
  int sA = m0 + ldRow;
  int aRow = -1;
  if (sA < cnt) aRow = rowMap ? rowMap[base + sA] : (base + sA);
  const ushort_t* aSrcRow = (aRow >= 0) ? (A + (size_t)aRow * lda) : A;

  // B staging: thread -> (one n column, 16 consecutive k)
  const int nn  = tid & 127;           // n within tile
  const int kph = tid >> 7;            // 0/1: low/high 16 k
  const int nB  = n0 + nn;

  auto stageA = [&](int k0, ushort_t* bufBase) {
    ushort_t* dst = bufBase + ldRow * 32 + ldCol;
    if (aRow >= 0) {
      if (k0 + ldCol + 15 < K) {
        const ushort_t* src = aSrcRow + k0 + ldCol;
        unsigned d0 = (unsigned)(uintptr_t)dst;
        unsigned d1 = (unsigned)(uintptr_t)(dst + 8);
        asm volatile("global_load_async_to_lds_b128 %0, %1, off"
                     :: "v"(d0), "v"(src) : "memory");
        asm volatile("global_load_async_to_lds_b128 %0, %1, off"
                     :: "v"(d1), "v"(src + 8) : "memory");
      } else {
#pragma unroll
        for (int j = 0; j < 16; ++j) {
          int k = k0 + ldCol + j;
          dst[j] = (k < K) ? aSrcRow[k] : (ushort_t)0;
        }
      }
    } else {
      U128 z = {0u, 0u, 0u, 0u};
      *(U128*)dst = z;
      *(U128*)(dst + 8) = z;
    }
  };

  auto loadB = [&](int k0, float* bre) {
    const int kb = k0 + kph * 16;
    if (nB < N && kb + 15 < K) {                 // fast unpredicated path
      const float* src = Bp + (size_t)kb * ldb + nB;
#pragma unroll
      for (int j = 0; j < 16; ++j) bre[j] = src[(size_t)j * ldb];
    } else {
#pragma unroll
      for (int j = 0; j < 16; ++j)
        bre[j] = (nB < N && (kb + j) < K) ? Bp[(size_t)(kb + j) * ldb + nB] : 0.f;
    }
  };
  auto storeB = [&](ushort_t* bufBase, const float* bre) {
    uint_t* dst = (uint_t*)(bufBase + nn * 32 + kph * 16);
#pragma unroll
    for (int r = 0; r < 8; ++r) dst[r] = pack2(bre[2 * r], bre[2 * r + 1]);
  };

  const int kTiles = (K + 31) >> 5;
  float bre[16];

  // prologue: tile 0 into buffer 0
  stageA(0, Al[0]);
  loadB(0, bre);
  storeB(Bl[0], bre);
  asm volatile("s_wait_asynccnt 0" ::: "memory");
  __syncthreads();

  for (int kt = 0; kt < kTiles; ++kt) {
    const int p = kt & 1;
    ushort_t* Ap = Al[p];
    ushort_t* Bc = Bl[p];
    const bool more = (kt + 1) < kTiles;
    if (more) {                       // kick next tile's loads early
      stageA((kt + 1) << 5, Al[1 - p]);
      loadB((kt + 1) << 5, bre);
    }

    v16bf af[2];
#pragma unroll
    for (int mi = 0; mi < 2; ++mi) {
      const U128* q = (const U128*)&Ap[(wm * 32 + mi * 16 + ln) * 32];
      FragU f; f.q[0] = q[half]; f.q[1] = q[2 + half];
      af[mi] = f.v;
    }
    v16bf bfr[4];
#pragma unroll
    for (int ni = 0; ni < 4; ++ni) {
      const U128* q = (const U128*)&Bc[(wn * 64 + ni * 16 + ln) * 32];
      FragU f; f.q[0] = q[half]; f.q[1] = q[2 + half];
      bfr[ni] = f.v;
    }
#pragma unroll
    for (int mi = 0; mi < 2; ++mi)
#pragma unroll
      for (int ni = 0; ni < 4; ++ni)
        acc[mi][ni] = __builtin_amdgcn_wmma_f32_16x16x32_bf16(
            false, af[mi], false, bfr[ni], (short)0, acc[mi][ni], false, false);

    if (more) {
      storeB(Bl[1 - p], bre);         // convert after the WMMAs (overlap loads)
      asm volatile("s_wait_asynccnt 0" ::: "memory");
    }
    __syncthreads();
  }

#pragma unroll
  for (int mi = 0; mi < 2; ++mi) {
#pragma unroll
    for (int ni = 0; ni < 4; ++ni) {
      int col = n0 + wn * 64 + ni * 16 + ln;
      if (col >= N) continue;
#pragma unroll
      for (int r = 0; r < 8; ++r) {
        int ml = m0 + wm * 32 + mi * 16 + r + 8 * half;
        if (ml < cnt) {
          size_t o = (size_t)(base + ml) * ldc + col;
          float v = acc[mi][ni][r];
          C[o] = addToC ? (C[o] + v) : v;
        }
      }
    }
  }
}

// ---------------- RoPE + bf16 convert + V transpose ----------------------
__global__ __launch_bounds__(256)
void rope_kernel(const float* __restrict__ qf, const float* __restrict__ kf,
                 const float* __restrict__ vf,
                 ushort_t* __restrict__ qb, ushort_t* __restrict__ kb,
                 ushort_t* __restrict__ vT, int T) {
  int n = blockIdx.x;
  int t = n % T;
  int b = n / T;
  for (int p = threadIdx.x; p < 512; p += 256) {    // q: 16 heads * 32 pairs
    int i = p & 31;
    float fr = __powf(10000.f, -(float)i * (1.f / 32.f));
    float ang = (float)t * fr;
    float sf, cf; __sincosf(ang, &sf, &cf);
    size_t o = (size_t)n * 1024 + (size_t)p * 2;
    float r = qf[o], im = qf[o + 1];
    *(uint_t*)&qb[o] = pack2((r * cf - im * sf) * 0.125f,   // fold 1/sqrt(64)
                             (r * sf + im * cf) * 0.125f);
  }
  for (int p = threadIdx.x; p < 256; p += 256) {    // k: 8 heads * 32 pairs
    int i = p & 31;
    float fr = __powf(10000.f, -(float)i * (1.f / 32.f));
    float ang = (float)t * fr;
    float sf, cf; __sincosf(ang, &sf, &cf);
    size_t o = (size_t)n * 512 + (size_t)p * 2;
    float r = kf[o], im = kf[o + 1];
    *(uint_t*)&kb[o] = pack2(r * cf - im * sf, r * sf + im * cf);
  }
  for (int epos = threadIdx.x; epos < 512; epos += 256) { // vT[b][h][d][t]
    int hh = epos >> 6, d = epos & 63;
    vT[(((size_t)b * 8 + hh) * 64 + d) * T + t] = f2b(vf[(size_t)n * 512 + epos]);
  }
}

// ---------------- flash attention: 1 wave per (b,h,16-q-row tile) ---------
__global__ __launch_bounds__(32)
void attn_kernel(const ushort_t* __restrict__ qb, const ushort_t* __restrict__ kb,
                 const ushort_t* __restrict__ vT, ushort_t* __restrict__ yb, int T) {
  const int bh = blockIdx.x;
  const int b = bh >> 4, h = bh & 15;
  const int hkv = h >> 1;                 // n_rep = 2
  const int q0 = blockIdx.y * 16;
  const int lane = threadIdx.x;
  const int ln = lane & 15;
  const int half = lane >> 4;

  __shared__ __align__(16) ushort_t Pl[16 * 32];

  v8f zero = {0.f,0.f,0.f,0.f,0.f,0.f,0.f,0.f};
  v16bf qfrag[2];
  {
    const ushort_t* qrow = qb + (size_t)(b * T + q0 + ln) * 1024 + h * 64;
#pragma unroll
    for (int c = 0; c < 2; ++c) {
      const U128* p = (const U128*)(qrow + c * 32);
      FragU f; f.q[0] = p[half]; f.q[1] = p[2 + half];
      qfrag[c] = f.v;
    }
  }

  v8f O[4];
#pragma unroll
  for (int ni = 0; ni < 4; ++ni) O[ni] = zero;
  float mprev[8], lsum[8];
#pragma unroll
  for (int r = 0; r < 8; ++r) { mprev[r] = -1e30f; lsum[r] = 0.f; }

  const ushort_t* vbase = vT + ((size_t)b * 8 + hkv) * 64 * T;
  const int kend = q0 + 16;
  for (int kb0 = 0; kb0 < kend; kb0 += 32) {
    v8f S[2];
#pragma unroll
    for (int j = 0; j < 2; ++j) {
      S[j] = zero;
      const ushort_t* krow = kb + (size_t)(b * T + kb0 + j * 16 + ln) * 512 + hkv * 64;
#pragma unroll
      for (int c = 0; c < 2; ++c) {
        const U128* p = (const U128*)(krow + c * 32);
        FragU f; f.q[0] = p[half]; f.q[1] = p[2 + half];
        S[j] = __builtin_amdgcn_wmma_f32_16x16x32_bf16(
            false, qfrag[c], false, f.v, (short)0, S[j], false, false);
      }
    }
#pragma unroll
    for (int r = 0; r < 8; ++r) {
      int qpos = q0 + r + 8 * half;
      float s0 = ((kb0 + ln)      <= qpos) ? S[0][r] : -1e30f;
      float s1 = ((kb0 + 16 + ln) <= qpos) ? S[1][r] : -1e30f;
      float mx = fmaxf(s0, s1);
#pragma unroll
      for (int d = 1; d < 16; d <<= 1) mx = fmaxf(mx, __shfl_xor(mx, d, 32));
      float mnew = fmaxf(mprev[r], mx);
      float alpha = __expf(mprev[r] - mnew);
      float p0 = __expf(s0 - mnew);
      float p1 = __expf(s1 - mnew);
      float ls = p0 + p1;
#pragma unroll
      for (int d = 1; d < 16; d <<= 1) ls += __shfl_xor(ls, d, 32);
      lsum[r] = lsum[r] * alpha + ls;
      mprev[r] = mnew;
#pragma unroll
      for (int ni = 0; ni < 4; ++ni) O[ni][r] *= alpha;
      int row = r + 8 * half;
      Pl[row * 32 + ln]      = f2b(p0);
      Pl[row * 32 + 16 + ln] = f2b(p1);
    }
    __syncthreads();
    v16bf pf;
    {
      const U128* p = (const U128*)&Pl[ln * 32];
      FragU f; f.q[0] = p[half]; f.q[1] = p[2 + half];
      pf = f.v;
    }
#pragma unroll
    for (int ni = 0; ni < 4; ++ni) {
      const ushort_t* vrow = vbase + (size_t)(ni * 16 + ln) * T + kb0;
      const U128* p = (const U128*)vrow;
      FragU f; f.q[0] = p[half]; f.q[1] = p[2 + half];
      O[ni] = __builtin_amdgcn_wmma_f32_16x16x32_bf16(
          false, pf, false, f.v, (short)0, O[ni], false, false);
    }
    __syncthreads();
  }
#pragma unroll
  for (int r = 0; r < 8; ++r) {
    float inv = 1.f / lsum[r];
    ushort_t* yrow = yb + (size_t)(b * T + q0 + r + 8 * half) * 1024 + h * 64;
#pragma unroll
    for (int ni = 0; ni < 4; ++ni) yrow[ni * 16 + ln] = f2b(O[ni][r] * inv);
  }
}

// ---------------- MoE gate, top-2, routing, silu*mul, combine -------------
__global__ __launch_bounds__(256)
void gate_kernel(const ushort_t* __restrict__ hb, const float* __restrict__ gw,
                 float* __restrict__ gl) {
  int n = blockIdx.x;
  int e = threadIdx.x >> 5, lane = threadIdx.x & 31;
  float s = 0.f;
  for (int c = lane; c < 1024; c += 32)
    s += b2f(hb[(size_t)n * 1024 + c]) * gw[c * 8 + e];
#pragma unroll
  for (int d = 16; d; d >>= 1) s += __shfl_xor(s, d, 32);
  if (lane == 0) gl[n * 8 + e] = s;
}

__global__ __launch_bounds__(256)
void topk_kernel(const float* __restrict__ gl, int* __restrict__ sel,
                 float* __restrict__ wts, int Ntok) {
  int n = blockIdx.x * blockDim.x + threadIdx.x;
  if (n >= Ntok) return;
  float g[8];
#pragma unroll
  for (int e = 0; e < 8; ++e) g[e] = gl[n * 8 + e];
  int i0 = 0;
#pragma unroll
  for (int e = 1; e < 8; ++e) if (g[e] > g[i0]) i0 = e;
  int i1 = (i0 == 0) ? 1 : 0;
#pragma unroll
  for (int e = 0; e < 8; ++e) if (e != i0 && g[e] > g[i1]) i1 = e;
  float e1 = __expf(g[i1] - g[i0]);
  float w0 = 1.f / (1.f + e1);
  sel[n * 2] = i0; sel[n * 2 + 1] = i1;
  wts[n * 2] = w0; wts[n * 2 + 1] = e1 * w0;
}

__global__ void route_kernel(const int* __restrict__ sel, int* __restrict__ slotTok,
                             int* __restrict__ tokSlot, int* __restrict__ eCnt,
                             int* __restrict__ eOff, int Ntok) {
  __shared__ int cs[8], os[8];
  int t = threadIdx.x;
  if (t < 8) {
    int c = 0;
    for (int n = 0; n < Ntok; ++n) {
      if (sel[n * 2] == t) c++;
      if (sel[n * 2 + 1] == t) c++;
    }
    cs[t] = c;
  }
  __syncthreads();
  if (t == 0) { int o = 0; for (int e = 0; e < 8; ++e) { os[e] = o; o += cs[e]; } }
  __syncthreads();
  if (t < 8) {
    eCnt[t] = cs[t];
    eOff[t] = os[t];
    int pos = os[t];
    for (int n = 0; n < Ntok; ++n) {
      if (sel[n * 2] == t)     { slotTok[pos] = n; tokSlot[n * 2] = pos; pos++; }
      if (sel[n * 2 + 1] == t) { slotTok[pos] = n; tokSlot[n * 2 + 1] = pos; pos++; }
    }
  }
}

__global__ __launch_bounds__(256)
void silu_mul_kernel(const float* __restrict__ g1, const float* __restrict__ g3,
                     ushort_t* __restrict__ act, long long total) {
  long long i = (long long)blockIdx.x * 256 + threadIdx.x;
  if (i >= total) return;
  float a = g1[i];
  float s = a / (1.f + __expf(-a));
  act[i] = f2b(s * g3[i]);
}

__global__ __launch_bounds__(256)
void combine_kernel(float* __restrict__ x, const float* __restrict__ eo,
                    const int* __restrict__ tokSlot, const float* __restrict__ wts) {
  int n = blockIdx.x;
  int s0 = tokSlot[n * 2], s1 = tokSlot[n * 2 + 1];
  float w0 = wts[n * 2], w1 = wts[n * 2 + 1];
  for (int c = threadIdx.x; c < 1024; c += 256)
    x[(size_t)n * 1024 + c] += w0 * eo[(size_t)s0 * 1024 + c]
                             + w1 * eo[(size_t)s1 * 1024 + c];
}

// ---------------- tied lm_head: out[b][v] = xf[b] . wte[v] ----------------
__global__ __launch_bounds__(256)
void lmhead_kernel(const float* __restrict__ xf, const float* __restrict__ wte,
                   float* __restrict__ out, int V) {
  int v = blockIdx.x, b = blockIdx.y;
  const float* wr = wte + (size_t)v * 1024;
  const float* xr = xf + (size_t)b * 1024;
  float s = 0.f;
  for (int c = threadIdx.x; c < 1024; c += 256) s += xr[c] * wr[c];
  s = blockReduceSum256(s);
  if (threadIdx.x == 0) out[(size_t)b * V + v] = s;
}

// --------------------------------------------------------------------------
extern "C" void kernel_launch(void* const* d_in, const int* in_sizes, int n_in,
                              void* d_out, int out_size, void* d_ws, size_t ws_size,
                              hipStream_t stream) {
  (void)in_sizes; (void)n_in; (void)out_size; (void)ws_size;
  const int*   idx = (const int*)d_in[0];
  const float* wte = (const float*)d_in[1];
  const float* ln1 = (const float*)d_in[2];
  const float* wq  = (const float*)d_in[3];
  const float* wk  = (const float*)d_in[4];
  const float* wv  = (const float*)d_in[5];
  const float* wo  = (const float*)d_in[6];
  const float* ln2 = (const float*)d_in[7];
  const float* gw  = (const float*)d_in[8];
  const float* w1  = (const float*)d_in[9];
  const float* w2  = (const float*)d_in[10];
  const float* w3  = (const float*)d_in[11];
  const float* lnf = (const float*)d_in[12];
  float* out = (float*)d_out;

  constexpr int Bc = 2, Tc = 1024, Cc = 1024, Vc = 32000, HIDc = 2728;
  constexpr int Nt = Bc * Tc;          // 2048 tokens
  constexpr int Sl = Nt * 2;           // 4096 routed slots (top-2)

  char* p = (char*)d_ws;
  auto carve = [&](size_t bytes) -> char* {
    char* r = p; p += (bytes + 255) & ~(size_t)255; return r;
  };
  float*    x    = (float*)carve((size_t)Nt * Cc * 4);
  float*    qf   = (float*)carve((size_t)Nt * 1024 * 4);
  float*    kf   = (float*)carve((size_t)Nt * 512 * 4);
  float*    vf   = (float*)carve((size_t)Nt * 512 * 4);
  ushort_t* hb   = (ushort_t*)carve((size_t)Nt * 1024 * 2);
  ushort_t* qb2  = (ushort_t*)carve((size_t)Nt * 1024 * 2);
  ushort_t* kb2  = (ushort_t*)carve((size_t)Nt * 512 * 2);
  ushort_t* vtb  = (ushort_t*)carve((size_t)Bc * 8 * 64 * Tc * 2);
  ushort_t* yb   = (ushort_t*)carve((size_t)Nt * 1024 * 2);
  float*    gl   = (float*)carve((size_t)Nt * 8 * 4);
  int*      sel  = (int*)carve((size_t)Nt * 2 * 4);
  float*    wtsb = (float*)carve((size_t)Nt * 2 * 4);
  int*  slotTok  = (int*)carve((size_t)Sl * 4);
  int*  tokSlot  = (int*)carve((size_t)Nt * 2 * 4);
  int*      eCnt = (int*)carve(8 * 4);
  int*      eOff = (int*)carve(8 * 4);
  float*    g1   = (float*)carve((size_t)Sl * HIDc * 4);
  float*    g3   = (float*)carve((size_t)Sl * HIDc * 4);
  ushort_t* actb = (ushort_t*)carve((size_t)Sl * HIDc * 2);
  float*    eo   = (float*)carve((size_t)Sl * Cc * 4);
  float*    xf   = (float*)carve((size_t)Bc * Cc * 4);

  embed_kernel<<<Nt, 256, 0, stream>>>(idx, wte, x);

  for (int l = 0; l < 2; ++l) {
    // ---- attention ----
    rmsnorm_kernel<<<Nt, 256, 0, stream>>>(x, ln1 + l * Cc, hb, nullptr, 1, 0);
    gemm_bf16_kernel<<<dim3(16, 8, 1), 256, 0, stream>>>(
        hb, 1024, nullptr, wq + (size_t)l * Cc * 1024, 1024, 0,
        qf, 1024, 0, Nt, 1024, 1024, nullptr, nullptr);
    gemm_bf16_kernel<<<dim3(16, 4, 1), 256, 0, stream>>>(
        hb, 1024, nullptr, wk + (size_t)l * Cc * 512, 512, 0,
        kf, 512, 0, Nt, 512, 1024, nullptr, nullptr);
    gemm_bf16_kernel<<<dim3(16, 4, 1), 256, 0, stream>>>(
        hb, 1024, nullptr, wv + (size_t)l * Cc * 512, 512, 0,
        vf, 512, 0, Nt, 512, 1024, nullptr, nullptr);
    rope_kernel<<<Nt, 256, 0, stream>>>(qf, kf, vf, qb2, kb2, vtb, Tc);
    attn_kernel<<<dim3(Bc * 16, Tc / 16), 32, 0, stream>>>(qb2, kb2, vtb, yb, Tc);
    gemm_bf16_kernel<<<dim3(16, 8, 1), 256, 0, stream>>>(
        yb, 1024, nullptr, wo + (size_t)l * 1024 * Cc, 1024, 0,
        x, 1024, /*add=*/1, Nt, 1024, 1024, nullptr, nullptr);

    // ---- MoE ----
    rmsnorm_kernel<<<Nt, 256, 0, stream>>>(x, ln2 + l * Cc, hb, nullptr, 1, 0);
    gate_kernel<<<Nt, 256, 0, stream>>>(hb, gw + (size_t)l * Cc * 8, gl);
    topk_kernel<<<(Nt + 255) / 256, 256, 0, stream>>>(gl, sel, wtsb, Nt);
    route_kernel<<<1, 32, 0, stream>>>(sel, slotTok, tokSlot, eCnt, eOff, Nt);
    gemm_bf16_kernel<<<dim3(16, (HIDc + 127) / 128, 8), 256, 0, stream>>>(
        hb, 1024, slotTok, w1 + (size_t)l * 8 * Cc * HIDc, HIDc,
        (long long)Cc * HIDc, g1, HIDc, 0, Nt, HIDc, 1024, eOff, eCnt);
    gemm_bf16_kernel<<<dim3(16, (HIDc + 127) / 128, 8), 256, 0, stream>>>(
        hb, 1024, slotTok, w3 + (size_t)l * 8 * Cc * HIDc, HIDc,
        (long long)Cc * HIDc, g3, HIDc, 0, Nt, HIDc, 1024, eOff, eCnt);
    long long tot = (long long)Sl * HIDc;
    silu_mul_kernel<<<(unsigned)((tot + 255) / 256), 256, 0, stream>>>(g1, g3, actb, tot);
    gemm_bf16_kernel<<<dim3(16, 8, 8), 256, 0, stream>>>(
        actb, HIDc, nullptr, w2 + (size_t)l * 8 * HIDc * Cc, 1024,
        (long long)HIDc * Cc, eo, 1024, 0, Nt, 1024, HIDc, eOff, eCnt);
    combine_kernel<<<Nt, 256, 0, stream>>>(x, eo, tokSlot, wtsb);
  }

  // final rmsnorm on the two last-position rows only, then tied lm_head
  rmsnorm_kernel<<<Bc, 256, 0, stream>>>(x, lnf, nullptr, xf, Tc, Tc - 1);
  lmhead_kernel<<<dim3(Vc, Bc), 256, 0, stream>>>(xf, wte, out, Vc);
}
